// MultiHeadedAttention_29076928594259
// MI455X (gfx1250) — matrix operand order repaired and checked by reading, not compiled, not run on hardware
//
#include <hip/hip_runtime.h>
#include <hip/hip_bf16.h>

// ---------------------------------------------------------------------------
// MultiHeadedAttention on MI455X (gfx1250), wave32, WMMA f16->f32.
//
// Pipeline (all GEMMs via v_wmma_f32_16x16x32_f16):
//   1) convert query/key/value fp32 -> f16; transpose+convert Wq/Wk/Wv/Wo.
//   2) Qp = q @ Wq + bq (f16 out), Kp = k @ Wk + bk (f16), Vt = (v @ Wv + bv)
//      written per-(b,h)-group TRANSPOSED [HD][S] so PV's B operand is
//      K-contiguous.
//   3) per chunk of 8 groups: scores = Qp_g @ Kp_g^T (f32, 16 MB/group),
//      column logsumexp L[k] (softmax is over the QUERY axis!), then
//      X_g = exp(s - L[k]) @ V_g with on-the-fly f16 conversion in LDS.
//   4) out = X @ Wo + bo (f32).
//
// The generic GEMM streams its tiles with GLOBAL_LOAD_ASYNC_TO_LDS_B128 into
// double-buffered LDS (ASYNCcnt-tracked, s_wait_asynccnt), overlapping the
// next tile's DMA with the current tile's WMMAs.
// ---------------------------------------------------------------------------

typedef __attribute__((ext_vector_type(16))) _Float16 v16h;
typedef __attribute__((ext_vector_type(8)))  _Float16 v8h;
typedef __attribute__((ext_vector_type(4)))  _Float16 v4h;
typedef __attribute__((ext_vector_type(8)))  float    v8f;

static constexpr int kB  = 4;
static constexpr int kS  = 2048;
static constexpr int kD  = 1024;
static constexpr int kH  = 8;
static constexpr int kHD = 128;
static constexpr int kG  = kB * kH;            // 32 (b,h) groups
static constexpr int kM  = kB * kS;            // 8192 rows of activations
static constexpr size_t kACT = (size_t)kM * kD;      // 8,388,608 elements
static constexpr size_t kGRP = (size_t)kS * kHD;     // 262,144 elems per group
static constexpr int kGC = 8;                  // groups per scores chunk

// LDS tile row pitch in f16 (32 data + 8 pad keeps 16B alignment, breaks
// the 4-row bank-conflict period).
static constexpr int LP = 40;

// ---------------------------------------------------------------------------
// Async Global -> LDS (CDNA5): per-lane 16B transfer, ASYNCcnt-tracked.
// LDS address = low 32 bits of the generic pointer (aperture truncation).
// ---------------------------------------------------------------------------
__device__ inline void async_load_b128(unsigned lds_addr, const void* gaddr) {
  asm volatile("global_load_async_to_lds_b128 %0, %1, off"
               :: "v"(lds_addr), "v"(gaddr)
               : "memory");
}
__device__ inline void wait_async_le(int) {}  // (doc helper; real waits below)
#define S_WAIT_ASYNCCNT(n) asm volatile("s_wait_asynccnt " #n ::: "memory")

// ---------------------------------------------------------------------------
// fp32 -> f16 vector convert
// ---------------------------------------------------------------------------
__global__ __launch_bounds__(256) void cvt_f32_to_f16(const float* __restrict__ in,
                                                      _Float16* __restrict__ out,
                                                      int n4) {
  int i = blockIdx.x * 256 + threadIdx.x;
  if (i < n4) {
    const float4 v = ((const float4*)in)[i];
    v4h o;
    o.x = (_Float16)v.x; o.y = (_Float16)v.y;
    o.z = (_Float16)v.z; o.w = (_Float16)v.w;
    ((v4h*)out)[i] = o;
  }
}

// ---------------------------------------------------------------------------
// W[k][n] fp32 (1024x1024) -> Wt[n][k] f16 (LDS tiled transpose)
// ---------------------------------------------------------------------------
__global__ __launch_bounds__(256) void w_transpose_f16(const float* __restrict__ W,
                                                       _Float16* __restrict__ Wt) {
  __shared__ float tile[32][33];
  const int bx = blockIdx.x * 32;   // original column block (n)
  const int by = blockIdx.y * 32;   // original row block (k)
  const int tx = threadIdx.x & 31;
  const int ty = threadIdx.x >> 5;  // 0..7
#pragma unroll
  for (int j = 0; j < 4; ++j)
    tile[ty + j * 8][tx] = W[(size_t)(by + ty + j * 8) * kD + bx + tx];
  __syncthreads();
#pragma unroll
  for (int j = 0; j < 4; ++j)
    Wt[(size_t)(bx + ty + j * 8) * kD + by + tx] = (_Float16)tile[tx][ty + j * 8];
}

// ---------------------------------------------------------------------------
// Fragment helpers (CDNA5 16x16x32 f16 WMMA VGPR layouts, wave32).
// A (16x32): lanes 0-15 hold K{0..7,16..23}, lanes 16-31 hold K{8..15,24..31}.
// B (32x16): same K order as A (lanes 0-15: K{0..7,16..23}, 16-31: K{8..15,24..31}).
// ---------------------------------------------------------------------------
__device__ inline v16h load_a_frag(const _Float16* rowp, int hs) {
  v8h lo = *(const v8h*)(rowp + hs * 8);
  v8h hi = *(const v8h*)(rowp + 16 + hs * 8);
  return __builtin_shufflevector(lo, hi, 0,1,2,3,4,5,6,7,8,9,10,11,12,13,14,15);
}
__device__ inline v16h load_b_frag(const _Float16* colp, int hs) {
  v8h lo = *(const v8h*)(colp + hs * 8);
  v8h hi = *(const v8h*)(colp + 16 + hs * 8);
  return __builtin_shufflevector(lo, hi, 0,1,2,3,4,5,6,7,8,9,10,11,12,13,14,15);
}

// ---------------------------------------------------------------------------
// Generic WMMA GEMM: C[MxN] = A[MxK] @ Bt[NxK]^T (+bias).
// 256 threads = 8 waves (4 row x 2 col); wave patch 32x64 = 2x4 frags.
// Double-buffered LDS fed by async global->LDS DMA.
// EPI: 0 = f16 flat, 1 = f16 V-transpose scatter, 2 = f32 flat.
// ---------------------------------------------------------------------------
template <int EPI>
__global__ __launch_bounds__(256) void gemm_wmma_f16(
    const _Float16* __restrict__ A, const _Float16* __restrict__ Bt,
    const float* __restrict__ bias, void* __restrict__ Cout,
    int M, int N, int K, size_t sA, size_t sB, size_t sC) {
  A  += (size_t)blockIdx.z * sA;
  Bt += (size_t)blockIdx.z * sB;

  __shared__ __align__(16) _Float16 As[2][128][LP];
  __shared__ __align__(16) _Float16 Bs[2][128][LP];

  const int t    = threadIdx.x;
  const int m0   = blockIdx.y * 128;
  const int n0   = blockIdx.x * 128;
  const int wid  = t >> 5, lane = t & 31;
  const int wrow = wid & 3, wcol = wid >> 2;
  const int r15  = lane & 15, hs = lane >> 4;

  v8f acc[2][4] = {};

  // This thread's slice of the tile DMA: rows (r, r+64), 8-f16 chunk c8.
  const int r    = t >> 2;          // 0..63
  const int c8   = (t & 3) * 8;     // 0,8,16,24
  const unsigned aBase0 = (unsigned)(size_t)&As[0][0][0];
  const unsigned aBase1 = (unsigned)(size_t)&As[1][0][0];
  const unsigned bBase0 = (unsigned)(size_t)&Bs[0][0][0];
  const unsigned bBase1 = (unsigned)(size_t)&Bs[1][0][0];
  const unsigned offLo  = (unsigned)((r * LP + c8) * 2);
  const unsigned offHi  = (unsigned)(((r + 64) * LP + c8) * 2);

  auto issue_tile = [&](int buf, int k0) {
    const unsigned ab = buf ? aBase1 : aBase0;
    const unsigned bb = buf ? bBase1 : bBase0;
    async_load_b128(ab + offLo, &A [(size_t)(m0 + r) * K + k0 + c8]);
    async_load_b128(ab + offHi, &A [(size_t)(m0 + r + 64) * K + k0 + c8]);
    async_load_b128(bb + offLo, &Bt[(size_t)(n0 + r) * K + k0 + c8]);
    async_load_b128(bb + offHi, &Bt[(size_t)(n0 + r + 64) * K + k0 + c8]);
  };

  const int nt = K / 32;
  issue_tile(0, 0);
  for (int i = 0; i < nt; ++i) {
    const int buf = i & 1;
    if (i + 1 < nt) {
      issue_tile(buf ^ 1, (i + 1) * 32);
      // 4 loads just issued may remain outstanding; tile i (issued earlier,
      // in-order completion) is guaranteed complete at <=4.
      S_WAIT_ASYNCCNT(0x4);
    } else {
      S_WAIT_ASYNCCNT(0x0);
    }
    __syncthreads();   // all waves' DMA for tile i visible in LDS

    v16h af[2], bf[4];
#pragma unroll
    for (int ri = 0; ri < 2; ++ri)
      af[ri] = load_a_frag(&As[buf][wrow * 32 + ri * 16 + r15][0], hs);
#pragma unroll
    for (int ci = 0; ci < 4; ++ci)
      bf[ci] = load_b_frag(&Bs[buf][wcol * 64 + ci * 16 + r15][0], hs);
#pragma unroll
    for (int ri = 0; ri < 2; ++ri)
#pragma unroll
      for (int ci = 0; ci < 4; ++ci)
        acc[ri][ci] = __builtin_amdgcn_wmma_f32_16x16x32_f16(
            false, af[ri], false, bf[ci], (short)0, acc[ri][ci], false, false);
    __syncthreads();   // buffer `buf` free before tile i+2 DMA targets it
  }

  // Epilogue. C frag layout: VGPR e, lane -> row = e + 8*(lane>=16), col = lane&15.
  const int rowo = hs * 8;
#pragma unroll
  for (int ri = 0; ri < 2; ++ri) {
#pragma unroll
    for (int ci = 0; ci < 4; ++ci) {
      const int gcol = n0 + wcol * 64 + ci * 16 + r15;
      const float bv = bias ? bias[gcol] : 0.0f;
#pragma unroll
      for (int e = 0; e < 8; ++e) {
        const int grow = m0 + wrow * 32 + ri * 16 + rowo + e;
        const float val = acc[ri][ci][e] + bv;
        if (EPI == 0) {
          ((_Float16*)Cout)[(size_t)blockIdx.z * sC + (size_t)grow * N + gcol] =
              (_Float16)val;
        } else if (EPI == 1) {
          // V-transpose scatter: flat idx -> group view [g][s2][d2] -> [g][d2][s2]
          const size_t idx = (size_t)grow * N + gcol;   // N == 1024 here
          const size_t g  = idx >> 18;                  // / 262144
          const size_t rr = idx & (kGRP - 1);
          const size_t s2 = rr >> 7, d2 = rr & 127;
          ((_Float16*)Cout)[g * kGRP + d2 * (size_t)kS + s2] = (_Float16)val;
        } else {
          ((float*)Cout)[(size_t)blockIdx.z * sC + (size_t)grow * N + gcol] = val;
        }
      }
    }
  }
}

// ---------------------------------------------------------------------------
// Column-wise logsumexp (softmax is over the QUERY axis => per key column).
// grid (kS/256, GC); thread t owns column k; rows read coalesced.
// ---------------------------------------------------------------------------
__global__ __launch_bounds__(256) void col_lse(const float* __restrict__ sc,
                                               float* __restrict__ lse) {
  const int gz = blockIdx.y;
  const float* s = sc + (size_t)gz * kS * kS;
  const int k = blockIdx.x * 256 + threadIdx.x;
  float m = -1e30f;
  for (int q = 0; q < kS; ++q) m = fmaxf(m, s[(size_t)q * kS + k]);
  float z = 0.0f;
  for (int q = 0; q < kS; ++q) z += __expf(s[(size_t)q * kS + k] - m);
  lse[(size_t)gz * kS + k] = m + __logf(z);
}

// ---------------------------------------------------------------------------
// PV GEMM: X_g[2048x128] = exp(scores - L[k]) @ V_g, WMMA with on-the-fly
// f16 conversion of the P tile in LDS.  grid (1, kS/128, GC).
// ---------------------------------------------------------------------------
__global__ __launch_bounds__(256) void pv_wmma(const float* __restrict__ sc,
                                               const float* __restrict__ lse,
                                               const _Float16* __restrict__ vt,
                                               _Float16* __restrict__ xh,
                                               int group_base) {
  __shared__ __align__(16) _Float16 As[128][LP];
  __shared__ __align__(16) _Float16 Bs[128][LP];

  const int gz = blockIdx.z;
  const int g  = group_base + gz;
  const float*    Sg = sc  + (size_t)gz * kS * kS;
  const float*    Lg = lse + (size_t)gz * kS;
  const _Float16* Vg = vt  + (size_t)g * kGRP;   // [HD][S], K-contiguous

  const int t    = threadIdx.x;
  const int m0   = blockIdx.y * 128;
  const int wid  = t >> 5, lane = t & 31;
  const int wrow = wid & 3, wcol = wid >> 2;
  const int r15  = lane & 15, hs = lane >> 4;

  v8f acc[2][4] = {};

  for (int k0 = 0; k0 < kS; k0 += 32) {
    // P tile: 128x32 f32 -> exp(s - L) -> f16
#pragma unroll
    for (int i = 0; i < 4; ++i) {
      const int idx = t + i * 256;        // 1024 float4 chunks
      const int r = idx >> 3, c4 = (idx & 7) * 4;
      const float4 v = *(const float4*)&Sg[(size_t)(m0 + r) * kS + k0 + c4];
      As[r][c4 + 0] = (_Float16)__expf(v.x - Lg[k0 + c4 + 0]);
      As[r][c4 + 1] = (_Float16)__expf(v.y - Lg[k0 + c4 + 1]);
      As[r][c4 + 2] = (_Float16)__expf(v.z - Lg[k0 + c4 + 2]);
      As[r][c4 + 3] = (_Float16)__expf(v.w - Lg[k0 + c4 + 3]);
    }
    // V tile: Bs[n=d2][kk] = Vg[n*kS + k0+kk]
#pragma unroll
    for (int i = 0; i < 2; ++i) {
      const int idx = t + i * 256;
      const int n = idx >> 2, c8 = (idx & 3) * 8;
      *(v8h*)&Bs[n][c8] = *(const v8h*)&Vg[(size_t)n * kS + k0 + c8];
    }
    if (k0 + 32 < kS)
      __builtin_prefetch(&Sg[(size_t)(m0 + (t >> 3)) * kS + k0 + 32], 0, 0);
    __syncthreads();

    v16h af[2], bf[4];
#pragma unroll
    for (int ri = 0; ri < 2; ++ri)
      af[ri] = load_a_frag(&As[wrow * 32 + ri * 16 + r15][0], hs);
#pragma unroll
    for (int ci = 0; ci < 4; ++ci)
      bf[ci] = load_b_frag(&Bs[wcol * 64 + ci * 16 + r15][0], hs);
#pragma unroll
    for (int ri = 0; ri < 2; ++ri)
#pragma unroll
      for (int ci = 0; ci < 4; ++ci)
        acc[ri][ci] = __builtin_amdgcn_wmma_f32_16x16x32_f16(
            false, af[ri], false, bf[ci], (short)0, acc[ri][ci], false, false);
    __syncthreads();
  }

  const int rowo = hs * 8;
#pragma unroll
  for (int ri = 0; ri < 2; ++ri)
#pragma unroll
    for (int ci = 0; ci < 4; ++ci) {
      const int gcol = wcol * 64 + ci * 16 + r15;   // d2 in [0,128)
#pragma unroll
      for (int e = 0; e < 8; ++e) {
        const int grow = m0 + wrow * 32 + ri * 16 + rowo + e;
        xh[(size_t)g * kGRP + (size_t)grow * kHD + gcol] =
            (_Float16)acc[ri][ci][e];
      }
    }
}

// ---------------------------------------------------------------------------
// Host orchestration
// ---------------------------------------------------------------------------
extern "C" void kernel_launch(void* const* d_in, const int* in_sizes, int n_in,
                              void* d_out, int out_size, void* d_ws, size_t ws_size,
                              hipStream_t stream) {
  const float* query = (const float*)d_in[0];
  const float* key   = (const float*)d_in[1];
  const float* value = (const float*)d_in[2];
  const float* Wq = (const float*)d_in[3];  const float* bq = (const float*)d_in[4];
  const float* Wk = (const float*)d_in[5];  const float* bk = (const float*)d_in[6];
  const float* Wv = (const float*)d_in[7];  const float* bv = (const float*)d_in[8];
  const float* Wo = (const float*)d_in[9];  const float* bo = (const float*)d_in[10];
  float* out = (float*)d_out;

  // Workspace layout (256B aligned)
  char* w = (char*)d_ws;
  size_t off = 0;
  auto take = [&](size_t bytes) {
    void* p = w + off;
    off = (off + bytes + 255) & ~(size_t)255;
    return p;
  };
  _Float16* qh  = (_Float16*)take(kACT * 2);
  _Float16* kh  = (_Float16*)take(kACT * 2);
  _Float16* vh  = (_Float16*)take(kACT * 2);
  _Float16* Wqt = (_Float16*)take((size_t)kD * kD * 2);
  _Float16* Wkt = (_Float16*)take((size_t)kD * kD * 2);
  _Float16* Wvt = (_Float16*)take((size_t)kD * kD * 2);
  _Float16* Wot = (_Float16*)take((size_t)kD * kD * 2);
  _Float16* Qp  = (_Float16*)take(kACT * 2);
  _Float16* Kp  = (_Float16*)take(kACT * 2);
  _Float16* Vt  = (_Float16*)take(kACT * 2);
  _Float16* Xh  = (_Float16*)take(kACT * 2);
  float*    Lse = (float*)take((size_t)kGC * kS * 4);
  float*    Scr = (float*)take((size_t)kGC * kS * kS * 4);   // 128 MB
  (void)ws_size; (void)in_sizes; (void)n_in; (void)out_size;

  // 1) precision conversion
  const int n4 = (int)(kACT / 4);
  cvt_f32_to_f16<<<dim3((n4 + 255) / 256), 256, 0, stream>>>(query, qh, n4);
  cvt_f32_to_f16<<<dim3((n4 + 255) / 256), 256, 0, stream>>>(key,   kh, n4);
  cvt_f32_to_f16<<<dim3((n4 + 255) / 256), 256, 0, stream>>>(value, vh, n4);
  w_transpose_f16<<<dim3(32, 32), 256, 0, stream>>>(Wq, Wqt);
  w_transpose_f16<<<dim3(32, 32), 256, 0, stream>>>(Wk, Wkt);
  w_transpose_f16<<<dim3(32, 32), 256, 0, stream>>>(Wv, Wvt);
  w_transpose_f16<<<dim3(32, 32), 256, 0, stream>>>(Wo, Wot);

  // 2) QKV projections [8192x1024] = act @ Wt^T + b
  dim3 gproj(kD / 128, kM / 128, 1);
  gemm_wmma_f16<0><<<gproj, 256, 0, stream>>>(qh, Wqt, bq, Qp, kM, kD, kD, 0, 0, 0);
  gemm_wmma_f16<0><<<gproj, 256, 0, stream>>>(kh, Wkt, bk, Kp, kM, kD, kD, 0, 0, 0);
  gemm_wmma_f16<1><<<gproj, 256, 0, stream>>>(vh, Wvt, bv, Vt, kM, kD, kD, 0, 0, 0);

  // 3) attention per chunk of kGC groups
  for (int c = 0; c < kG / kGC; ++c) {
    const int gbase = c * kGC;
    // scores = Q_g @ K_g^T  [2048x2048] f32, batched over kGC groups
    dim3 gsc(kS / 128, kS / 128, kGC);
    gemm_wmma_f16<2><<<gsc, 256, 0, stream>>>(
        Qp + (size_t)gbase * kGRP, Kp + (size_t)gbase * kGRP, nullptr, Scr,
        kS, kS, kHD, kGRP, kGRP, (size_t)kS * kS);
    // per-column logsumexp (softmax over query axis)
    col_lse<<<dim3(kS / 256, kGC), 256, 0, stream>>>(Scr, Lse);
    // X_g = softmax_cols(scores) @ V_g
    pv_wmma<<<dim3(1, kS / 128, kGC), 256, 0, stream>>>(Scr, Lse, Vt, Xh, gbase);
  }

  // 4) output projection: out = X @ Wo + bo (f32)
  gemm_wmma_f16<2><<<gproj, 256, 0, stream>>>(Xh, Wot, bo, out, kM, kD, kD, 0, 0, 0);
}
